// GraphAttentionLayer_88270167867414
// MI455X (gfx1250) — matrix operand order repaired and checked by reading, not compiled, not run on hardware
//
#include <hip/hip_runtime.h>
#include <hip/hip_bf16.h>
#include <math.h>

typedef __attribute__((ext_vector_type(16))) _Float16 v16h;
typedef __attribute__((ext_vector_type(8)))  _Float16 v8h;
typedef __attribute__((ext_vector_type(8)))  float    v8f;
typedef __attribute__((ext_vector_type(4)))  float    v4f;

union V16U { v16h v; v8h h8[2]; };

#define T_LEN 2048
#define NHEAD 4
#define DHEAD 64
#define BATCH 8
#define IN_DIM 256

static __device__ __forceinline__ v8f wmma_f16(v16h a, v16h b, v8f c) {
    return __builtin_amdgcn_wmma_f32_16x16x32_f16(false, a, false, b, (short)0, c, false, false);
}

// Broadcast lane (j) -> lanes 0-15, lane (j+8) -> lanes 16-31 without LDS:
// two v_readlane_b32 + one v_cndmask.
static __device__ __forceinline__ float bcast_row(float v, int j, int half) {
    int lo = __builtin_amdgcn_readlane(__float_as_int(v), j);
    int hi = __builtin_amdgcn_readlane(__float_as_int(v), j + 8);
    return __int_as_float(half ? hi : lo);
}

// ---------------------------------------------------------------------------
// Kernel 0: W (256x256 f32, [in][out]) -> Wt16 (f16, [out][in])
// ---------------------------------------------------------------------------
__global__ __launch_bounds__(256) void gat_cvtW(const float* __restrict__ W,
                                                _Float16* __restrict__ Wt) {
    int i = blockIdx.x * 256 + threadIdx.x;          // 0..65535
    int in = i >> 8, out = i & 255;
    Wt[out * IN_DIM + in] = (_Float16)W[in * IN_DIM + out];
}

// ---------------------------------------------------------------------------
// Kernel 1: Wh = h @ W  via WMMA.  One wave per 16x16 output tile.
// Emits Wh16[(b*4+h), t, d] and WhT16[(b*4+h), d, t] in f16.
// ---------------------------------------------------------------------------
__global__ __launch_bounds__(128) void gat_proj(const float* __restrict__ h,
                                                const _Float16* __restrict__ Wt,
                                                _Float16* __restrict__ Wh,
                                                _Float16* __restrict__ WhT) {
    const int lane = threadIdx.x & 31;
    const int wave = threadIdx.x >> 5;
    const int tile = blockIdx.x * 4 + wave;          // 0..16383
    const int mt = tile >> 4;                        // row tile  0..1023
    const int nt = tile & 15;                        // col tile  0..15
    const int half = lane >> 4, nl = lane & 15;

    const int arow = mt * 16 + nl;
    v8f acc = (v8f)0.0f;

#pragma unroll
    for (int c = 0; c < 8; ++c) {                    // K = 256 = 8 x 32
        const float* ap = h + (size_t)arow * IN_DIM + c * 32 + half * 16;
        v4f f0 = *(const v4f*)(ap);
        v4f f1 = *(const v4f*)(ap + 4);
        v4f f2 = *(const v4f*)(ap + 8);
        v4f f3 = *(const v4f*)(ap + 12);
        V16U a;
#pragma unroll
        for (int e = 0; e < 4; ++e) {
            a.v[e]      = (_Float16)f0[e];
            a.v[4 + e]  = (_Float16)f1[e];
            a.v[8 + e]  = (_Float16)f2[e];
            a.v[12 + e] = (_Float16)f3[e];
        }
        v16h b = *(const v16h*)(Wt + (size_t)(nt * 16 + nl) * IN_DIM + c * 32 + half * 16);
        acc = wmma_f16(a.v, b, acc);
    }

#pragma unroll
    for (int j = 0; j < 8; ++j) {
        int r    = j + 8 * half;
        int grow = mt * 16 + r;                      // b*2048 + t
        int gcol = nt * 16 + nl;                     // head*64 + d
        int b_   = grow >> 11, t = grow & 2047;
        int hd   = gcol >> 6,  d = gcol & 63;
        int bh   = b_ * NHEAD + hd;
        _Float16 hv = (_Float16)acc[j];
        Wh [((size_t)bh * T_LEN + t) * DHEAD + d] = hv;
        WhT[((size_t)bh * DHEAD + d) * T_LEN + t] = hv;
    }
}

// ---------------------------------------------------------------------------
// Kernel 2: flash attention. One wave = 32 queries, key blocks of 32.
// Scores computed transposed (S^T = K @ Q^T) so the probability tile lands
// in the P@V A-operand layout with an in-lane pack (no LDS bounce).
// All 16 b128 loads issued up front per block; 16 WMMAs per block.
// ---------------------------------------------------------------------------
__global__ __launch_bounds__(128) void gat_attn(const _Float16* __restrict__ Wh,
                                                const _Float16* __restrict__ WhT,
                                                float* __restrict__ out) {
    const int lane = threadIdx.x & 31;
    const int wave = threadIdx.x >> 5;
    const int tile = blockIdx.x * 4 + wave;          // 0..2047
    const int qt = tile & 63;                        // 32-query tile in head
    const int bh = tile >> 6;                        // 0..31
    const int half = lane >> 4, nl = lane & 15;

    const _Float16* base  = Wh  + (size_t)bh * T_LEN * DHEAD;
    const _Float16* baseT = WhT + (size_t)bh * DHEAD * T_LEN;

    // Q resident as B-operand: qb[qgroup][dchunk]; lane nl = query column
    v16h qb[2][2];
#pragma unroll
    for (int g = 0; g < 2; ++g) {
        const _Float16* qp = base + (size_t)(qt * 32 + g * 16 + nl) * DHEAD;
        qb[g][0] = *(const v16h*)(qp + half * 16);
        qb[g][1] = *(const v16h*)(qp + 32 + half * 16);
    }

    float m[2]    = {-3.0e38f, -3.0e38f};
    float lsum[2] = {0.0f, 0.0f};
    v8f oacc[2][4];
#pragma unroll
    for (int g = 0; g < 2; ++g)
#pragma unroll
        for (int n = 0; n < 4; ++n) oacc[g][n] = (v8f)0.0f;

    const float scale = 0.125f;                      // 1/sqrt(64)

    for (int s0 = 0; s0 < T_LEN; s0 += 32) {
        // ---- issue ALL loads for this block: K tiles (A op) + V tiles (B op) ----
        V16U ka[2][2];
#pragma unroll
        for (int kt = 0; kt < 2; ++kt)
#pragma unroll
            for (int c = 0; c < 2; ++c) {
                const _Float16* kp = base + (size_t)(s0 + kt * 16 + nl) * DHEAD
                                   + c * 32 + half * 16;
                ka[kt][c].h8[0] = *(const v8h*)(kp);
                ka[kt][c].h8[1] = *(const v8h*)(kp + 8);
            }
        V16U vb[4];
#pragma unroll
        for (int n = 0; n < 4; ++n) {
            vb[n].h8[0] = *(const v8h*)(baseT + (size_t)(n * 16 + nl) * T_LEN + s0 + half * 8);
            vb[n].h8[1] = *(const v8h*)(baseT + (size_t)(n * 16 + nl) * T_LEN + s0 + 16 + half * 8);
        }

        // ---- S^T tiles: st[kt][g]: rows = keys, cols = queries ----
        v8f st[2][2];
#pragma unroll
        for (int kt = 0; kt < 2; ++kt)
#pragma unroll
            for (int g = 0; g < 2; ++g) {
                v8f s = (v8f)0.0f;
                s = wmma_f16(ka[kt][0].v, qb[g][0], s);
                s = wmma_f16(ka[kt][1].v, qb[g][1], s);
                st[kt][g] = s;
            }

        // ---- scale + LeakyReLU + online softmax (per lane: query g*16+nl,
        //      16 keys; other key-half lives in lane^16) ----
        V16U pa[2];
        float alpha[2];
#pragma unroll
        for (int g = 0; g < 2; ++g) {
            float p[16];
#pragma unroll
            for (int j = 0; j < 8; ++j) {
                float x0 = st[0][g][j] * scale; p[j]     = x0 > 0.0f ? x0 : 0.2f * x0;
                float x1 = st[1][g][j] * scale; p[8 + j] = x1 > 0.0f ? x1 : 0.2f * x1;
            }
            float mx = p[0];
#pragma unroll
            for (int e = 1; e < 16; ++e) mx = fmaxf(mx, p[e]);
            mx = fmaxf(mx, __shfl_xor(mx, 16));
            float mnew = fmaxf(m[g], mx);
            alpha[g] = __expf(m[g] - mnew);
            m[g] = mnew;
            float rs = 0.0f;
#pragma unroll
            for (int e = 0; e < 16; ++e) { p[e] = __expf(p[e] - mnew); rs += p[e]; }
            rs += __shfl_xor(rs, 16);
            lsum[g] = lsum[g] * alpha[g] + rs;
            // in-lane pack: C layout of S^T == A layout of P (16 queries x 32 keys)
#pragma unroll
            for (int e = 0; e < 16; ++e) pa[g].v[e] = (_Float16)p[e];
        }

        // ---- rescale O accumulators (readlane broadcast, no LDS) ----
#pragma unroll
        for (int j = 0; j < 8; ++j) {
            float a0 = bcast_row(alpha[0], j, half);
            float a1 = bcast_row(alpha[1], j, half);
#pragma unroll
            for (int n = 0; n < 4; ++n) { oacc[0][n][j] *= a0; oacc[1][n][j] *= a1; }
        }

        // ---- O += P @ V ----
#pragma unroll
        for (int n = 0; n < 4; ++n) {
            oacc[0][n] = wmma_f16(pa[0].v, vb[n].v, oacc[0][n]);
            oacc[1][n] = wmma_f16(pa[1].v, vb[n].v, oacc[1][n]);
        }
    }

    // ---- normalize and write out (B, T, 256) f32 ----
    const int b_ = bh >> 2, hd = bh & 3;
#pragma unroll
    for (int g = 0; g < 2; ++g)
#pragma unroll
        for (int j = 0; j < 8; ++j) {
            float inv = 1.0f / bcast_row(lsum[g], j, half);
            int t = qt * 32 + g * 16 + j + 8 * half;
            float* op = out + ((size_t)b_ * T_LEN + t) * (NHEAD * DHEAD) + hd * DHEAD + nl;
#pragma unroll
            for (int n = 0; n < 4; ++n) op[n * 16] = oacc[g][n][j] * inv;
        }
}

extern "C" void kernel_launch(void* const* d_in, const int* in_sizes, int n_in,
                              void* d_out, int out_size, void* d_ws, size_t ws_size,
                              hipStream_t stream) {
    const float* h = (const float*)d_in[0];          // (8, 2048, 256) f32
    const float* W = (const float*)d_in[1];          // (256, 256) f32
    float* out = (float*)d_out;                      // (8, 2048, 256) f32

    char* ws = (char*)d_ws;
    _Float16* Wt16  = (_Float16*)ws;                                   // 128 KB
    _Float16* Wh16  = (_Float16*)(ws + (128u << 10));                  // 8 MB
    _Float16* WhT16 = (_Float16*)(ws + (128u << 10) + (size_t)BATCH * NHEAD * T_LEN * DHEAD * 2);

    gat_cvtW<<<256, 256, 0, stream>>>(W, Wt16);
    gat_proj<<<4096, 128, 0, stream>>>(h, Wt16, Wh16, WhT16);          // 16384 tiles / 4 waves
    gat_attn<<<512, 128, 0, stream>>>(Wh16, WhT16, out);               // 2048 tiles / 4 waves
}
